// SpeechAutoencoder_27616639713439
// MI455X (gfx1250) — matrix operand order repaired and checked by reading, not compiled, not run on hardware
//
#include <hip/hip_runtime.h>

// ---------------------------------------------------------------------------
// Types for CDNA5 WMMA bf16: A/B = v16bf (8 VGPRs of packed bf16), C/D = v8f
// ---------------------------------------------------------------------------
typedef __attribute__((ext_vector_type(16))) __bf16 v16bf;
typedef __attribute__((ext_vector_type(8)))  __bf16 v8bf;
typedef __attribute__((ext_vector_type(8)))  float  v8f;

#define WMMA_BF16(a, b, c)                                                    \
  __builtin_amdgcn_wmma_f32_16x16x32_bf16(false, (a), false, (b), (short)0,   \
                                          (c), false, false)

static __device__ __forceinline__ __bf16 f2bf(float f) {
  union { float f; unsigned u; } in;
  in.f = f;
  unsigned r = in.u + 0x7FFFu + ((in.u >> 16) & 1u);  // round-to-nearest-even
  unsigned short s = (unsigned short)(r >> 16);
  return __builtin_bit_cast(__bf16, s);
}

// A-matrix fragment (16x32 bf16) loaded straight from a row-major bf16 source.
// Layout: lanes 0-15 are rows M=0..15 holding K 0..15;
//         lanes 16-31 same rows holding K 16..31 (matches B).
// -> two 16-byte loads per lane, no conversion VALU.
static __device__ __forceinline__ v16bf load_a_bf16(
    const __bf16* __restrict__ src, int ld, int m0, int k0, int lane) {
  const int half = lane >> 4, l = lane & 15;
  const __bf16* row = src + (size_t)(m0 + l) * ld + k0 + half * 16;
  union { v16bf v; struct { v8bf lo; v8bf hi; } p; } u;
  u.p.lo = *reinterpret_cast<const v8bf*>(row);        // K base+0..7
  u.p.hi = *reinterpret_cast<const v8bf*>(row + 8);    // K base+8..15
  return u.v;
}

// B-matrix fragment (32x16 bf16) from row-major bf16 weights W[N, K]:
// B(k, n) = W[n, k]. Lane = column n; lanes 0-15 hold K 0..15, lanes 16-31
// hold K 16..31 -> one contiguous 32-byte read of a weight row.
static __device__ __forceinline__ v16bf load_b_bf16(
    const __bf16* __restrict__ W, int ldw, int n0, int k0, int lane) {
  const int half = lane >> 4, l = lane & 15;
  return *reinterpret_cast<const v16bf*>(W + (size_t)(n0 + l) * ldw + k0 +
                                         half * 16);
}

// ---------------------------------------------------------------------------
// One GRU cell step for a 32x16 (batch x hidden) tile per wave: two M-tiles
// share every B (weight) fragment -> halves L2 weight traffic per step.
// A operands (x, h) are read as bf16; h_prev for the z-blend is read fp32.
// Writes h_out in fp32 (recurrence accuracy) and bf16 (next step's A input).
// ---------------------------------------------------------------------------
static __device__ __forceinline__ void gru_tile(
    const __bf16* __restrict__ xb, int ldx, int Kx,
    const float* __restrict__ hin_f, const __bf16* __restrict__ hin_b,
    float* __restrict__ hout_f, __bf16* __restrict__ hout_b,
    const __bf16* __restrict__ Wih, const __bf16* __restrict__ Whh,
    const float* __restrict__ bih, const float* __restrict__ bhh) {
  const int H = 1024;
  const int wave = (blockIdx.x * blockDim.x + threadIdx.x) >> 5;
  const int lane = threadIdx.x & 31;
  const int n_tiles = H / 16;  // 64
  if (wave >= 2 * n_tiles) return;  // 2 M-groups of 32 rows each
  const int m0 = (wave / n_tiles) * 32;
  const int n0 = (wave % n_tiles) * 16;

  v8f ar0 = {}, ar1 = {}, az0 = {}, az1 = {};
  v8f ai0 = {}, ai1 = {}, ah0 = {}, ah1 = {};

  // Input contribution: x @ Wih.T (gate rows: r=[0,H), z=[H,2H), n=[2H,3H))
#pragma clang loop unroll(disable)
  for (int k0 = 0; k0 < Kx; k0 += 32) {
    v16bf a0 = load_a_bf16(xb, ldx, m0,      k0, lane);
    v16bf a1 = load_a_bf16(xb, ldx, m0 + 16, k0, lane);
    v16bf br = load_b_bf16(Wih, Kx, n0,         k0, lane);
    v16bf bz = load_b_bf16(Wih, Kx, n0 + H,     k0, lane);
    v16bf bn = load_b_bf16(Wih, Kx, n0 + 2 * H, k0, lane);
    ar0 = WMMA_BF16(a0, br, ar0);  ar1 = WMMA_BF16(a1, br, ar1);
    az0 = WMMA_BF16(a0, bz, az0);  az1 = WMMA_BF16(a1, bz, az1);
    ai0 = WMMA_BF16(a0, bn, ai0);  ai1 = WMMA_BF16(a1, bn, ai1);
  }
  // Hidden contribution: h @ Whh.T (n-gate kept separate for r*hn)
#pragma clang loop unroll(disable)
  for (int k0 = 0; k0 < H; k0 += 32) {
    v16bf a0 = load_a_bf16(hin_b, H, m0,      k0, lane);
    v16bf a1 = load_a_bf16(hin_b, H, m0 + 16, k0, lane);
    v16bf br = load_b_bf16(Whh, H, n0,         k0, lane);
    v16bf bz = load_b_bf16(Whh, H, n0 + H,     k0, lane);
    v16bf bn = load_b_bf16(Whh, H, n0 + 2 * H, k0, lane);
    ar0 = WMMA_BF16(a0, br, ar0);  ar1 = WMMA_BF16(a1, br, ar1);
    az0 = WMMA_BF16(a0, bz, az0);  az1 = WMMA_BF16(a1, bz, az1);
    ah0 = WMMA_BF16(a0, bn, ah0);  ah1 = WMMA_BF16(a1, bn, ah1);
  }

  const int half = lane >> 4, l = lane & 15;
  const int n = n0 + l;
  const float b_r  = bih[n] + bhh[n];
  const float b_z  = bih[H + n] + bhh[H + n];
  const float b_in = bih[2 * H + n];
  const float b_hn = bhh[2 * H + n];
#pragma unroll
  for (int g = 0; g < 2; ++g) {
    const v8f& vr = g ? ar1 : ar0;
    const v8f& vz = g ? az1 : az0;
    const v8f& vi = g ? ai1 : ai0;
    const v8f& vh = g ? ah1 : ah0;
#pragma unroll
    for (int e = 0; e < 8; ++e) {
      const int m = m0 + g * 16 + e + 8 * half;
      const float hprev = hin_f[(size_t)m * H + n];
      const float r  = 1.0f / (1.0f + __expf(-(vr[e] + b_r)));
      const float z  = 1.0f / (1.0f + __expf(-(vz[e] + b_z)));
      const float nn = tanhf(vi[e] + b_in + r * (vh[e] + b_hn));
      const float hnew = (1.0f - z) * nn + z * hprev;
      hout_f[(size_t)m * H + n] = hnew;
      hout_b[(size_t)m * H + n] = f2bf(hnew);
    }
  }
}

// Encoder round: blockIdx.y==0 -> layer0 step t; blockIdx.y==1 -> layer1 step
// t-1 (independent jobs within a round, issued in one launch).
__global__ void __launch_bounds__(256, 1) enc_round(
    const __bf16* x0, int ldx0, const float* h0inf, const __bf16* h0inb,
    float* h0outf, __bf16* h0outb, const __bf16* Wih0, const __bf16* Whh0,
    const float* bih0, const float* bhh0, int do0,
    const __bf16* x1, const float* h1inf, const __bf16* h1inb, float* h1outf,
    __bf16* h1outb, const __bf16* Wih1, const __bf16* Whh1, const float* bih1,
    const float* bhh1, int do1) {
  if (blockIdx.y == 0) {
    if (!do0) return;
    gru_tile(x0, ldx0, 160, h0inf, h0inb, h0outf, h0outb, Wih0, Whh0, bih0,
             bhh0);
  } else {
    if (!do1) return;
    gru_tile(x1, 1024, 1024, h1inf, h1inb, h1outf, h1outb, Wih1, Whh1, bih1,
             bhh1);
  }
}

// Decoder GRUCell step: input is the compact bf16 buffer written by proj_step
__global__ void __launch_bounds__(256, 1) dec_gru_step(
    const __bf16* xb, const float* hinf, const __bf16* hinb, float* houtf,
    __bf16* houtb, const __bf16* Wih, const __bf16* Whh, const float* bih,
    const float* bhh) {
  gru_tile(xb, 160, 160, hinf, hinb, houtf, houtb, Wih, Whh, bih, bhh);
}

// fc: h = tanh(concat(hA, hB) @ fcW.T + fcb); fcW is [H, 2H] row-major
__global__ void __launch_bounds__(256, 1) fc_tanh(
    const __bf16* __restrict__ hA, const __bf16* __restrict__ hB,
    const __bf16* __restrict__ W, const float* __restrict__ bias,
    float* __restrict__ outf, __bf16* __restrict__ outb) {
  const int H = 1024;
  const int wave = (blockIdx.x * blockDim.x + threadIdx.x) >> 5;
  const int lane = threadIdx.x & 31;
  const int n_tiles = H / 16;
  if (wave >= 4 * n_tiles) return;
  const int m0 = (wave / n_tiles) * 16;
  const int n0 = (wave % n_tiles) * 16;
  v8f acc = {};
#pragma clang loop unroll(disable)
  for (int k0 = 0; k0 < H; k0 += 32) {
    v16bf a = load_a_bf16(hA, H, m0, k0, lane);
    v16bf b = load_b_bf16(W, 2 * H, n0, k0, lane);
    acc = WMMA_BF16(a, b, acc);
  }
#pragma clang loop unroll(disable)
  for (int k0 = 0; k0 < H; k0 += 32) {
    v16bf a = load_a_bf16(hB, H, m0, k0, lane);
    v16bf b = load_b_bf16(W, 2 * H, n0, H + k0, lane);
    acc = WMMA_BF16(a, b, acc);
  }
  const int half = lane >> 4, l = lane & 15, n = n0 + l;
  const float bb = bias[n];
#pragma unroll
  for (int e = 0; e < 8; ++e) {
    const float v = tanhf(acc[e] + bb);
    outf[(size_t)(m0 + e + 8 * half) * H + n] = v;
    outb[(size_t)(m0 + e + 8 * half) * H + n] = f2bf(v);
  }
}

// Projection: out[:, t, :] = hd @ pW.T + pb  (M=64, N=160, K=1024).
// 32x16 tile per wave (B fragment reused by 2 WMMAs). Also writes a compact
// bf16 copy consumed by the next decoder GRU step.
__global__ void __launch_bounds__(256, 1) proj_step(
    const __bf16* __restrict__ hdb, const __bf16* __restrict__ pW,
    const float* __restrict__ pb, float* __restrict__ out, int ldo,
    __bf16* __restrict__ ibf) {
  const int H = 1024, Fo = 160;
  const int wave = (blockIdx.x * blockDim.x + threadIdx.x) >> 5;
  const int lane = threadIdx.x & 31;
  const int n_tiles = Fo / 16;  // 10
  if (wave >= 2 * n_tiles) return;  // 20 waves
  const int m0 = (wave / n_tiles) * 32;
  const int n0 = (wave % n_tiles) * 16;
  v8f acc0 = {}, acc1 = {};
#pragma clang loop unroll(disable)
  for (int k0 = 0; k0 < H; k0 += 32) {
    v16bf a0 = load_a_bf16(hdb, H, m0,      k0, lane);
    v16bf a1 = load_a_bf16(hdb, H, m0 + 16, k0, lane);
    v16bf b  = load_b_bf16(pW, H, n0, k0, lane);
    acc0 = WMMA_BF16(a0, b, acc0);
    acc1 = WMMA_BF16(a1, b, acc1);
  }
  const int half = lane >> 4, l = lane & 15, n = n0 + l;
  const float bb = pb[n];
#pragma unroll
  for (int g = 0; g < 2; ++g) {
    const v8f& av = g ? acc1 : acc0;
#pragma unroll
    for (int e = 0; e < 8; ++e) {
      const int m = m0 + g * 16 + e + 8 * half;
      const float v = av[e] + bb;
      out[(size_t)m * ldo + n] = v;
      ibf[(size_t)m * Fo + n] = f2bf(v);
    }
  }
}

// fp32 -> bf16 conversion (weights + encoder input; one-time per launch)
__global__ void __launch_bounds__(256) f32_to_bf16(
    const float* __restrict__ src, __bf16* __restrict__ dst, int n) {
  int i = blockIdx.x * blockDim.x + threadIdx.x;
  if (i < n) dst[i] = f2bf(src[i]);
}

// ---------------------------------------------------------------------------
extern "C" void kernel_launch(void* const* d_in, const int* in_sizes, int n_in,
                              void* d_out, int out_size, void* d_ws,
                              size_t ws_size, hipStream_t stream) {
  (void)in_sizes; (void)n_in; (void)out_size; (void)ws_size;
  const int B_ = 64, T_ = 1024, F_ = 160, H_ = 1024;
  const int BH = B_ * H_;

  const float* x     = (const float*)d_in[0];
  const float* Wih0  = (const float*)d_in[1];
  const float* Whh0  = (const float*)d_in[2];
  const float* bih0  = (const float*)d_in[3];
  const float* bhh0  = (const float*)d_in[4];
  const float* Wih1  = (const float*)d_in[5];
  const float* Whh1  = (const float*)d_in[6];
  const float* bih1  = (const float*)d_in[7];
  const float* bhh1  = (const float*)d_in[8];
  const float* fcW   = (const float*)d_in[9];
  const float* fcb   = (const float*)d_in[10];
  const float* dWih  = (const float*)d_in[11];
  const float* dWhh  = (const float*)d_in[12];
  const float* dbih  = (const float*)d_in[13];
  const float* dbhh  = (const float*)d_in[14];
  const float* pW    = (const float*)d_in[15];
  const float* pb    = (const float*)d_in[16];
  float* out = (float*)d_out;  // [B, T, F]

  // Workspace: fp32 state ping-pongs first, then all bf16 arrays (16B-aligned:
  // every element count below is a multiple of 8).
  float* ws  = (float*)d_ws;
  float* h0s = ws;              // [2][B*H] encoder layer0 state (fp32)
  float* h1s = h0s + 2 * BH;    // [2][B*H] encoder layer1 state (fp32)
  float* hds = h1s + 2 * BH;    // [2][B*H] decoder state (fp32)
  __bf16* bp = (__bf16*)(hds + 2 * BH);
  __bf16* h0b = bp;             bp += 2 * BH;      // bf16 mirrors of states
  __bf16* h1b = bp;             bp += 2 * BH;
  __bf16* hdb = bp;             bp += 2 * BH;
  __bf16* ibf = bp;             bp += B_ * F_;     // decoder input (bf16)
  __bf16* bx  = bp;             bp += B_ * T_ * F_; // encoder input (bf16)
  __bf16* bWih0 = bp;           bp += 3 * H_ * F_;
  __bf16* bWhh0 = bp;           bp += 3 * H_ * H_;
  __bf16* bWih1 = bp;           bp += 3 * H_ * H_;
  __bf16* bWhh1 = bp;           bp += 3 * H_ * H_;
  __bf16* bdWih = bp;           bp += 3 * H_ * F_;
  __bf16* bdWhh = bp;           bp += 3 * H_ * H_;
  __bf16* bfcW  = bp;           bp += H_ * 2 * H_;
  __bf16* bpW   = bp;           bp += F_ * H_;

  // 1) One-time conversions: weights + full encoder input to bf16
  struct { const float* s; __bf16* d; int n; } conv[9] = {
      {x,    bx,    B_ * T_ * F_},
      {Wih0, bWih0, 3 * H_ * F_}, {Whh0, bWhh0, 3 * H_ * H_},
      {Wih1, bWih1, 3 * H_ * H_}, {Whh1, bWhh1, 3 * H_ * H_},
      {dWih, bdWih, 3 * H_ * F_}, {dWhh, bdWhh, 3 * H_ * H_},
      {fcW,  bfcW,  H_ * 2 * H_}, {pW,   bpW,   F_ * H_}};
  for (int i = 0; i < 9; ++i)
    f32_to_bf16<<<(conv[i].n + 255) / 256, 256, 0, stream>>>(conv[i].s,
                                                             conv[i].d,
                                                             conv[i].n);

  // 2) Zero initial states (state at time t lives in buffer t&1; t=-1 -> 1)
  hipMemsetAsync(h0s + BH, 0, (size_t)BH * sizeof(float), stream);
  hipMemsetAsync(h1s + BH, 0, (size_t)BH * sizeof(float), stream);
  hipMemsetAsync(h0b + BH, 0, (size_t)BH * sizeof(__bf16), stream);
  hipMemsetAsync(h1b + BH, 0, (size_t)BH * sizeof(__bf16), stream);
  hipMemsetAsync(ibf, 0, (size_t)B_ * F_ * sizeof(__bf16), stream);

  // 3) Encoder: layer0 step t + layer1 step t-1 fused per round
  // (128 waves per job -> 16 blocks of 8 waves)
  for (int t = 0; t <= T_; ++t) {
    const int do0 = (t < T_), do1 = (t >= 1);
    const int tt = (t >= 1) ? (t - 1) : 0;
    const int p0r = (t + 1) & 1, p0w = t & 1;      // layer0 read/write parity
    const int p1r = (tt + 1) & 1, p1w = tt & 1;    // layer1 read/write parity
    enc_round<<<dim3(16, 2), 256, 0, stream>>>(
        bx + (size_t)t * F_, T_ * F_, h0s + p0r * BH, h0b + p0r * BH,
        h0s + p0w * BH, h0b + p0w * BH, bWih0, bWhh0, bih0, bhh0, do0,
        h0b + p1w * BH /* = h0 state at time tt */, h1s + p1r * BH,
        h1b + p1r * BH, h1s + p1w * BH, h1b + p1w * BH, bWih1, bWhh1, bih1,
        bhh1, do1);
  }

  // 4) fc: hd_{-1} = tanh(concat(h0_final, h1_final) @ fcW.T + fcb)
  const int fin = (T_ - 1) & 1;
  fc_tanh<<<32, 256, 0, stream>>>(h0b + fin * BH, h1b + fin * BH, bfcW, fcb,
                                  hds + 1 * BH, hdb + 1 * BH);

  // 5) Autoregressive decoder: gru reads ibf (zeros at t=0, proj output after)
  for (int t = 0; t < T_; ++t) {
    const int pr = (t + 1) & 1, pw = t & 1;
    dec_gru_step<<<16, 256, 0, stream>>>(ibf, hds + pr * BH, hdb + pr * BH,
                                         hds + pw * BH, hdb + pw * BH, bdWih,
                                         bdWhh, dbih, dbhh);
    proj_step<<<3, 256, 0, stream>>>(hdb + pw * BH, bpW, pb,
                                     out + (size_t)t * F_, T_ * F_, ibf);
  }
}
